// LinearSplineLayer_28870770164370
// MI455X (gfx1250) — matrix operand order repaired and checked by reading, not compiled, not run on hardware
//
#include <hip/hip_runtime.h>
#include <hip/hip_bf16.h>

#define HDIM  512
#define KBINS 16
#define BATCH 8192
#define N2    (HDIM * KBINS)   // 8192
#define LDA   (HDIM + 8)       // padded LDS row stride (halves): distinct banks per lane
#define MSUB  4                // 4 x 16-row subtiles per wave -> 64 rows per block

typedef __attribute__((ext_vector_type(16))) _Float16 v16h;
typedef __attribute__((ext_vector_type(8)))  _Float16 v8h;
typedef __attribute__((ext_vector_type(4)))  _Float16 v4h;
typedef __attribute__((ext_vector_type(8)))  float    v8f;

// tanh via native exp: tanh(z) = 1 - 2/(e^{2z}+1); exact limits at +-inf.
__device__ __forceinline__ float tanh_fast(float z) {
    float e2 = __expf(2.0f * z);
    return 1.0f - 2.0f / (e2 + 1.0f);
}

// ---------------------------------------------------------------------------
// Transpose + f32->f16 convert: W[k][n] (rows x cols, row-major) -> Wt[n][k]
// ---------------------------------------------------------------------------
__global__ void transpose_w_kernel(const float* __restrict__ W,
                                   _Float16* __restrict__ Wt,
                                   int rows, int cols) {
    size_t total = (size_t)rows * cols;
    for (size_t i = (size_t)blockIdx.x * blockDim.x + threadIdx.x; i < total;
         i += (size_t)gridDim.x * blockDim.x) {
        size_t k = i / cols;
        size_t n = i % cols;
        Wt[n * (size_t)rows + k] = (_Float16)W[i];
    }
}

// ---------------------------------------------------------------------------
// Init output: v_out[:, :H] = v_passive ; log_density_out = log_density
// ---------------------------------------------------------------------------
__global__ void init_out_kernel(const float* __restrict__ v_in,
                                const float* __restrict__ log_density,
                                float* __restrict__ out) {
    size_t total = (size_t)BATCH * HDIM + BATCH;
    for (size_t i = (size_t)blockIdx.x * blockDim.x + threadIdx.x; i < total;
         i += (size_t)gridDim.x * blockDim.x) {
        if (i < (size_t)BATCH * HDIM) {
            size_t m = i >> 9;
            size_t c = i & (HDIM - 1);
            out[m * (2 * HDIM) + c] = v_in[m * (2 * HDIM) + c];
        } else {
            size_t m = i - (size_t)BATCH * HDIM;
            out[(size_t)BATCH * (2 * HDIM) + m] = log_density[m];
        }
    }
}

// ---------------------------------------------------------------------------
// GEMM1: h = tanh((v_passive - 0.5) @ W1 + b1), stored f16.
// 8 waves/block, one 16x16 tile per wave. grid = (HDIM/128, BATCH/16)
// ---------------------------------------------------------------------------
__global__ __launch_bounds__(256) void gemm1_kernel(
    const float* __restrict__ v_in, const float* __restrict__ b1,
    const _Float16* __restrict__ W1t, _Float16* __restrict__ Hh) {
    __shared__ _Float16 As[16 * LDA];

    const int t  = threadIdx.x;
    const int m0 = blockIdx.y * 16;

    // Stage A = (v_passive - 0.5) as f16: float4 loads, 8-byte LDS stores.
    // 16 rows x 512 cols = 2048 float4 chunks; chunks never cross a row.
    for (int i = 0; i < 8; ++i) {
        int fidx = i * 256 + t;             // float4 index 0..2047
        int fb   = fidx * 4;                // first float index
        int r    = fb >> 9;
        int c    = fb & (HDIM - 1);
        float4 x = *(const float4*)(v_in + (size_t)(m0 + r) * (2 * HDIM) + c);
        v4h h;
        h[0] = (_Float16)(x.x - 0.5f);
        h[1] = (_Float16)(x.y - 0.5f);
        h[2] = (_Float16)(x.z - 0.5f);
        h[3] = (_Float16)(x.w - 0.5f);
        *(v4h*)(&As[r * LDA + c]) = h;
    }
    __syncthreads();

    const int wave   = t >> 5;
    const int lane   = t & 31;
    const int half16 = lane >> 4;
    const int ln     = lane & 15;
    const int n      = blockIdx.x * 128 + wave * 16 + ln;

    const _Float16* arow = &As[ln * LDA];
    v8f acc = {};
    for (int k0 = 0; k0 < HDIM; k0 += 32) {
        v8h alo = *(const v8h*)(arow + k0 + half16 * 16);
        v8h ahi = *(const v8h*)(arow + k0 + 8 + half16 * 16);
        v16h a = __builtin_shufflevector(alo, ahi, 0, 1, 2, 3, 4, 5, 6, 7, 8,
                                         9, 10, 11, 12, 13, 14, 15);
        const _Float16* bptr = W1t + (size_t)n * HDIM + k0 + half16 * 16;
        v16h b = *(const v16h*)bptr;
        acc = __builtin_amdgcn_wmma_f32_16x16x32_f16(
            false, a, false, b, (short)0, acc, false, false);
    }

    const float bias  = b1[n];
    const int   mbase = m0 + half16 * 8;
    for (int r = 0; r < 8; ++r) {
        float y = tanh_fast(acc[r] + bias);
        Hh[(size_t)(mbase + r) * HDIM + n] = (_Float16)y;
    }
}

// ---------------------------------------------------------------------------
// GEMM2 fused with spline epilogue. Block = 256 threads = 8 waves, one hidden
// unit (hidx) per wave, M=64 rows per block (MSUB=4 accumulators per wave so
// each B fragment feeds 4 WMMAs). grid = (HDIM/8, BATCH/64)
// ---------------------------------------------------------------------------
__global__ __launch_bounds__(256) void gemm2_spline_kernel(
    const _Float16* __restrict__ Hh, const _Float16* __restrict__ W2t,
    const float* __restrict__ b2, const float* __restrict__ v_in,
    float* __restrict__ out) {
    __shared__ _Float16 As[64 * LDA];   // 64 rows x 512 K (padded), ~65 KB
    __shared__ float    logacc[64];     // per-row -sum(log p_k) partials

    const int t  = threadIdx.x;
    const int m0 = blockIdx.y * 64;

    if (t < 64) logacc[t] = 0.0f;

    // Stage 32x512 f16 rows (0..31) of h: b128 global loads -> b128 LDS stores.
    // 16384 halves = 2048 uint4 chunks; 8 halves never cross a 512-half row.
    {
        const uint4* src = (const uint4*)(Hh + (size_t)m0 * HDIM);
        for (int i = 0; i < 8; ++i) {
            int vidx = i * 256 + t;         // uint4 index 0..2047
            int hb   = vidx * 8;            // first half index
            int r    = hb >> 9;
            int c    = hb & (HDIM - 1);
            *(uint4*)(&As[r * LDA + c]) = src[vidx];
        }
    }
    __syncthreads();

    const int wave   = t >> 5;
    const int lane   = t & 31;
    const int half16 = lane >> 4;
    const int ln     = lane & 15;
    const int hidx   = blockIdx.x * 8 + wave;   // hidden unit 0..511
    const int n      = hidx * KBINS + ln;       // GEMM2 column; bin = ln
    const int base   = half16 * 16;             // first lane of this half

    v8f acc[MSUB] = {};
    for (int k0 = 0; k0 < HDIM; k0 += 32) {
        // One B fragment reused by MSUB WMMAs
        const _Float16* bptr = W2t + (size_t)n * HDIM + k0 + half16 * 16;
        v16h b = *(const v16h*)bptr;
        for (int s = 0; s < MSUB; ++s) {
            const _Float16* arow = &As[(s * 16 + ln) * LDA];
            v8h alo = *(const v8h*)(arow + k0 + half16 * 16);
            v8h ahi = *(const v8h*)(arow + k0 + 8 + half16 * 16);
            v16h a = __builtin_shufflevector(alo, ahi, 0, 1, 2, 3, 4, 5, 6, 7,
                                             8, 9, 10, 11, 12, 13, 14, 15);
            acc[s] = __builtin_amdgcn_wmma_f32_16x16x32_f16(
                false, a, false, b, (short)0, acc[s], false, false);
        }
    }

    const float bias = b2[n];
    for (int s = 0; s < MSUB; ++s) {
        const int mbase = m0 + s * 16 + half16 * 8;
        for (int r = 0; r < 8; ++r) {
            const int m = mbase + r;
            // tanh output in (-1,1) => softmax needs no max-subtraction.
            float th = tanh_fast(acc[s][r] + bias);
            float ex = __expf(th);             // in [1/e, e], always safe

            // inclusive prefix sum of ex within the 16-lane segment
            float pre = ex;
            for (int d = 1; d < 16; d <<= 1) {
                float v = __shfl_up(pre, d, 16);
                if (ln >= d) pre += v;
            }
            float excl = pre - ex;             // unnormalized knots_y

            // bin selection (searchsorted 'left' on uniform knots)
            float va = v_in[(size_t)m * (2 * HDIM) + HDIM + hidx];
            int kbin = (int)ceilf(va * (float)KBINS) - 1;
            kbin = max(0, min(KBINS - 1, kbin));

            // convergent shuffles: total sum = scan value at lane 15
            float stot = __shfl(pre, base + 15, 32);
            float exk  = __shfl(ex, base + kbin, 32);
            float prek = __shfl(excl, base + kbin, 32);

            if (ln == 0) {
                float rinv  = 1.0f / stot;     // single reciprocal
                float p_k   = exk * rinv;
                float y_lo  = prek * rinv;
                float alpha = va * (float)KBINS - (float)kbin;
                out[(size_t)m * (2 * HDIM) + HDIM + hidx] =
                    fmaf(alpha, p_k, y_lo);
                atomicAdd(&logacc[s * 16 + half16 * 8 + r], -__logf(p_k));
            }
        }
    }

    __syncthreads();
    if (t < 64)
        atomicAdd(&out[(size_t)BATCH * (2 * HDIM) + m0 + t], logacc[t]);
}

// ---------------------------------------------------------------------------
// Launch
// ---------------------------------------------------------------------------
extern "C" void kernel_launch(void* const* d_in, const int* in_sizes, int n_in,
                              void* d_out, int out_size, void* d_ws,
                              size_t ws_size, hipStream_t stream) {
    (void)in_sizes; (void)n_in; (void)out_size; (void)ws_size;
    const float* v_in        = (const float*)d_in[0];
    const float* log_density = (const float*)d_in[1];
    const float* W1          = (const float*)d_in[2];
    const float* b1          = (const float*)d_in[3];
    const float* W2          = (const float*)d_in[4];
    const float* b2          = (const float*)d_in[5];
    float* out = (float*)d_out;

    // Workspace layout (f16): W1t | W2t | Hh  (~17.3 MB total)
    char* ws = (char*)d_ws;
    _Float16* W1t = (_Float16*)(ws);
    _Float16* W2t = (_Float16*)(ws + (size_t)HDIM * HDIM * 2);
    _Float16* Hh  = (_Float16*)(ws + (size_t)HDIM * HDIM * 2 +
                                (size_t)HDIM * N2 * 2);

    transpose_w_kernel<<<1024, 256, 0, stream>>>(W1, W1t, HDIM, HDIM);
    transpose_w_kernel<<<4096, 256, 0, stream>>>(W2, W2t, HDIM, N2);

    init_out_kernel<<<4096, 256, 0, stream>>>(v_in, log_density, out);

    gemm1_kernel<<<dim3(HDIM / 128, BATCH / 16), 256, 0, stream>>>(
        v_in, b1, W1t, Hh);

    gemm2_spline_kernel<<<dim3(HDIM / 8, BATCH / 64), 256, 0, stream>>>(
        Hh, W2t, b2, v_in, out);
}
